// ImageTriplaneGenerator_15144054686222
// MI455X (gfx1250) — matrix operand order repaired and checked by reading, not compiled, not run on hardware
//
#include <hip/hip_runtime.h>
#include <hip/hip_bf16.h>

// ---------------------------------------------------------------------------
// MI455X / gfx1250 implementation of ImageTriplaneGenerator forward.
// Math refactor: Wk projection folded onto the query side and Wv projection
// applied to softmax-weighted context (valid since softmax weights sum to 1
// and bk*q is constant over l). GEMMs use v_wmma_f32_16x16x32_bf16 with a
// 2Mx4N register tile per wave32.
// ---------------------------------------------------------------------------

typedef __bf16 bf16;
typedef __attribute__((ext_vector_type(16))) __bf16 v16bf;
typedef __attribute__((ext_vector_type(8)))  __bf16 v8bf;
typedef __attribute__((ext_vector_type(8)))  float  v8f;

#define GSZ     64
#define SSAMP   4
#define CFEAT   771
#define NHEAD   3
#define NVIEW   4
#define QDIM    2313          // 3*771
#define KPAD    800           // 771 padded to mult of 32 (WMMA K)
#define NPADB   832           // 771 padded to mult of 64 (4 N-tiles per wave)
#define NPTS    4096          // G*G
#define LCTX    16            // S*NV
#define KCHUNKS 25            // KPAD/32
#define MGROUPS 128           // (NPTS/16)/2
#define NGROUPS 13            // (NPADB/16)/4

__device__ __forceinline__ bf16 f2bf(float x) {
  union { float f; unsigned u; } v; v.f = x;
  unsigned r = (v.u + 0x7FFFu + ((v.u >> 16) & 1u)) >> 16;
  unsigned short h = (unsigned short)r;
  return __builtin_bit_cast(bf16, h);
}
__device__ __forceinline__ float bf2f(bf16 b) {
  unsigned short h = __builtin_bit_cast(unsigned short, b);
  union { unsigned u; float f; } v; v.u = ((unsigned)h) << 16;
  return v.f;
}
__device__ __forceinline__ int iclamp63(int x) { return x < 0 ? 0 : (x > 63 ? 63 : x); }

__device__ __forceinline__ void plane_point(int plane, float ex, float ey, float es,
                                            float& p0, float& p1, float& p2) {
  if (plane == 0)      { p0 = ex; p1 = ey; p2 = es; }   // xy
  else if (plane == 1) { p0 = ex; p1 = es; p2 = ey; }   // xz
  else                 { p0 = es; p1 = ex; p2 = ey; }   // yz
}

// ---------------------------------------------------------------------------
// Weight prep: Wk slice per head as Bt[N=c][K=d] (already "transposed" view),
// Wv slice per head as Bt[N=dcol][K=c] (true transpose). bf16, zero padded
// to [NPADB x KPAD].
// ---------------------------------------------------------------------------
__global__ __launch_bounds__(256) void prep_wk_kernel(const float* __restrict__ Wk,
                                                      bf16* __restrict__ wkb) {
  int idx = blockIdx.x * 256 + threadIdx.x;
  int total = NHEAD * NPADB * KPAD;
  if (idx >= total) return;
  int h = idx / (NPADB * KPAD);
  int rem = idx - h * NPADB * KPAD;
  int c = rem / KPAD;
  int d = rem - c * KPAD;
  float v = (c < CFEAT && d < CFEAT) ? Wk[(size_t)c * QDIM + h * CFEAT + d] : 0.f;
  wkb[idx] = f2bf(v);
}

__global__ __launch_bounds__(256) void prep_wv_kernel(const float* __restrict__ Wv,
                                                      bf16* __restrict__ wvb) {
  int idx = blockIdx.x * 256 + threadIdx.x;
  int total = NHEAD * NPADB * KPAD;
  if (idx >= total) return;
  int h = idx / (NPADB * KPAD);
  int rem = idx - h * NPADB * KPAD;
  int dcol = rem / KPAD;
  int c = rem - dcol * KPAD;
  float v = (dcol < CFEAT && c < CFEAT) ? Wv[(size_t)c * QDIM + h * CFEAT + dcol] : 0.f;
  wvb[idx] = f2bf(v);
}

// ---------------------------------------------------------------------------
// One-time image feature transpose: [NV][C][4096] -> [NV][4096][C] so that
// the bilinear gather in sample_ctx reads contiguous channel rows (coalesced).
// LDS 32x32 tile transpose, bank-conflict padded.
// ---------------------------------------------------------------------------
__global__ __launch_bounds__(256) void transpose_imgf_kernel(const float* __restrict__ imgf,
                                                             float* __restrict__ imgfT) {
  __shared__ float tile[32][33];
  int m  = blockIdx.z;
  int pb = blockIdx.x * 32;              // pixel base
  int cb = blockIdx.y * 32;              // channel base
  int tx = threadIdx.x & 31;
  int ty = threadIdx.x >> 5;             // 0..7
  const float* src = imgf + (size_t)m * CFEAT * 4096;
  for (int i = ty; i < 32; i += 8) {
    int c = cb + i;
    tile[i][tx] = (c < CFEAT) ? src[(size_t)c * 4096 + pb + tx] : 0.f;
  }
  __syncthreads();
  float* dst = imgfT + (size_t)m * 4096 * CFEAT;
  for (int i = ty; i < 32; i += 8) {
    int c = cb + tx;
    if (c < CFEAT) dst[(size_t)(pb + i) * CFEAT + c] = tile[tx][i];
  }
}

// ---------------------------------------------------------------------------
// Context sampling: one block per (point n, context row l=(s,m)).
// Projects 3D point through w2c (analytic inverse of affine c2w) and K,
// bilinear-samples the TRANSPOSED image features (zeros padding), writes
// bf16 ctx row. Channel reads are contiguous.
// ---------------------------------------------------------------------------
__global__ __launch_bounds__(256) void sample_ctx_kernel(const float* __restrict__ imgfT,
                                                         const float* __restrict__ c2w,
                                                         const float* __restrict__ intr,
                                                         bf16* __restrict__ ctx, int plane) {
  int blk = blockIdx.x;        // n*16 + l
  int n = blk >> 4;
  int l = blk & 15;
  int s = l >> 2;
  int m = l & 3;
  int gxi = n & 63, gyi = n >> 6;
  float ex = -1.f + 2.f * (float)gxi / 63.f;
  float ey = -1.f + 2.f * (float)gyi / 63.f;
  float es = -1.f + 2.f * (float)s / 3.f;
  float p0, p1, p2;
  plane_point(plane, ex, ey, es, p0, p1, p2);

  const float* M = c2w + m * 16;           // c2w = [R|t; 0 1], w2c = [R^T | -R^T t]
  float dx = p0 - M[3], dy = p1 - M[7], dz = p2 - M[11];
  float cx = M[0] * dx + M[4] * dy + M[8]  * dz;
  float cy = M[1] * dx + M[5] * dy + M[9]  * dz;
  float cz = M[2] * dx + M[6] * dy + M[10] * dz;
  const float* Km = intr + m * 9;
  float ds = cz + 1e-8f;
  float u = (Km[0] * cx + Km[1] * cy + Km[2] * cz) / ds;
  float v = (Km[3] * cx + Km[4] * cy + Km[5] * cz) / ds;
  bool mask = (cz > 0.f) && (u >= 0.f) && (u < 64.f) && (v >= 0.f) && (v < 64.f);

  // grid-normalize then denormalize cancels: ix = u, iy = v  (align_corners)
  float x0f = floorf(u), y0f = floorf(v);
  float wx = u - x0f, wy = v - y0f;
  int x0 = (int)x0f, y0 = (int)y0f;
  int xs[4] = {x0, x0 + 1, x0, x0 + 1};
  int ys[4] = {y0, y0, y0 + 1, y0 + 1};
  float w4[4] = {(1.f - wx) * (1.f - wy), wx * (1.f - wy), (1.f - wx) * wy, wx * wy};
  const float* rows[4];
  float wgt[4];
  const float* base0 = imgfT + (size_t)m * 4096 * CFEAT;
#pragma unroll
  for (int k = 0; k < 4; ++k) {
    bool valid = (xs[k] >= 0) && (xs[k] <= 63) && (ys[k] >= 0) && (ys[k] <= 63);
    wgt[k] = (mask && valid) ? w4[k] : 0.f;
    rows[k] = base0 + (size_t)(iclamp63(ys[k]) * 64 + iclamp63(xs[k])) * CFEAT;
  }

  bf16* dst = ctx + ((size_t)n * LCTX + l) * KPAD;
  for (int c = threadIdx.x; c < KPAD; c += 256) {
    float val = 0.f;
    if (c < CFEAT)
      val = wgt[0] * rows[0][c] + wgt[1] * rows[1][c] + wgt[2] * rows[2][c] + wgt[3] * rows[3][c];
    dst[c] = f2bf(val);
  }
}

// ---------------------------------------------------------------------------
// Query assembly + LayerNorm. One block per point n. Builds the 2313-dim
// query (plane features + means of border-sampled other-plane features,
// faithful to the double grid-normalization in the reference), LayerNorms it,
// and writes per-head bf16 A-matrices [h][n][KPAD].
// ---------------------------------------------------------------------------
__global__ __launch_bounds__(256) void query_ln_kernel(const float* __restrict__ qxy,
                                                       const float* __restrict__ qxz,
                                                       const float* __restrict__ qyz,
                                                       const float* __restrict__ lng,
                                                       const float* __restrict__ lnb,
                                                       bf16* __restrict__ Aq, int plane) {
  __shared__ float qv[QDIM];
  __shared__ float red[256];
  __shared__ float sstat[2];
  int tid = threadIdx.x;
  int n = blockIdx.x;
  int gxi = n & 63, gyi = n >> 6;
  float ex = -1.f + 2.f * (float)gxi / 63.f;
  float ey = -1.f + 2.f * (float)gyi / 63.f;

  const float* qplanes[3] = {qxy, qxz, qyz};
  int o1, o2;
  if (plane == 0)      { o1 = 1; o2 = 2; }
  else if (plane == 1) { o1 = 0; o2 = 2; }
  else                 { o1 = 0; o2 = 1; }

  float wq[2][4][4];
  int   oq[2][4][4];
#pragma unroll
  for (int oi = 0; oi < 2; ++oi) {
    int op = (oi == 0) ? o1 : o2;
#pragma unroll
    for (int s = 0; s < 4; ++s) {
      float es = -1.f + 2.f * (float)s / 3.f;
      float p0, p1, p2;
      plane_point(plane, ex, ey, es, p0, p1, p2);
      float ca, cb;
      if (op == 0)      { ca = p0; cb = p1; }
      else if (op == 1) { ca = p0; cb = p2; }
      else              { ca = p1; cb = p2; }
      float g0 = fminf(fmaxf((ca * 0.5f + 0.5f) * 63.f, 0.f), 63.f);
      float g1 = fminf(fmaxf((cb * 0.5f + 0.5f) * 63.f, 0.f), 63.f);
      // faithful: pixel coords re-fed through normalized transform, border pad
      float ix = fminf(fmaxf((g0 + 1.f) * 0.5f * 63.f, 0.f), 63.f);
      float iy = fminf(fmaxf((g1 + 1.f) * 0.5f * 63.f, 0.f), 63.f);
      float x0f = floorf(ix), y0f = floorf(iy);
      float wx = ix - x0f, wy = iy - y0f;
      int x0 = (int)x0f, y0 = (int)y0f;
      int x1 = iclamp63(x0 + 1), y1 = iclamp63(y0 + 1);
      x0 = iclamp63(x0); y0 = iclamp63(y0);
      oq[oi][s][0] = y0 * 64 + x0; wq[oi][s][0] = (1.f - wx) * (1.f - wy);
      oq[oi][s][1] = y0 * 64 + x1; wq[oi][s][1] = wx * (1.f - wy);
      oq[oi][s][2] = y1 * 64 + x0; wq[oi][s][2] = (1.f - wx) * wy;
      oq[oi][s][3] = y1 * 64 + x1; wq[oi][s][3] = wx * wy;
    }
  }

  for (int c = tid; c < CFEAT; c += 256) {
    qv[c] = qplanes[plane][(size_t)n * CFEAT + c];
#pragma unroll
    for (int oi = 0; oi < 2; ++oi) {
      const float* qs = qplanes[(oi == 0) ? o1 : o2];
      float acc = 0.f;
#pragma unroll
      for (int s = 0; s < 4; ++s)
#pragma unroll
        for (int k = 0; k < 4; ++k)
          acc += wq[oi][s][k] * qs[(size_t)oq[oi][s][k] * CFEAT + c];
      qv[CFEAT + oi * CFEAT + c] = acc * 0.25f;
    }
  }
  __syncthreads();

  float ls = 0.f, lq = 0.f;
  for (int c = tid; c < QDIM; c += 256) { float t = qv[c]; ls += t; lq += t * t; }
  red[tid] = ls; __syncthreads();
  for (int st = 128; st > 0; st >>= 1) { if (tid < st) red[tid] += red[tid + st]; __syncthreads(); }
  if (tid == 0) sstat[0] = red[0];
  __syncthreads();
  red[tid] = lq; __syncthreads();
  for (int st = 128; st > 0; st >>= 1) { if (tid < st) red[tid] += red[tid + st]; __syncthreads(); }
  if (tid == 0) {
    float mu = sstat[0] / (float)QDIM;
    float var = red[0] / (float)QDIM - mu * mu;
    sstat[0] = mu;
    sstat[1] = rsqrtf(var + 1e-5f);
  }
  __syncthreads();
  float mu = sstat[0], rstd = sstat[1];

  for (int c = tid; c < QDIM; c += 256) {
    int h = c / CFEAT;
    int d = c - h * CFEAT;
    float vn = (qv[c] - mu) * rstd * lng[c] + lnb[c];
    Aq[((size_t)h * NPTS + n) * KPAD + d] = f2bf(vn);
  }
  for (int z = tid; z < NHEAD * (KPAD - CFEAT); z += 256) {
    int h = z / (KPAD - CFEAT);
    int d = CFEAT + (z - h * (KPAD - CFEAT));
    Aq[((size_t)h * NPTS + n) * KPAD + d] = f2bf(0.f);
  }
}

// ---------------------------------------------------------------------------
// WMMA GEMM, 2Mx4N register tile per wave32 (32x64 output per wave).
// C[M x Ncols] (+bias) = A[M x KPAD] * Bt[Ncols x KPAD]^T, bf16 in, f32 out.
// A fragment (ISA 7.12.2, 16-bit A 16x32): lane row = lane%16, half = lane/16,
//   elems 0..7 -> K = half*8 + e ; elems 8..15 -> K = 16 + half*8 + (e-8).
// B fragment (16-bit B 32x16): lane col = lane%16, same K order as A
//   (two 16B loads from the [N][K] Bt row).
// D: col = lane%16, VGPR r -> row r + 8*half.
// ---------------------------------------------------------------------------
__device__ __forceinline__ v16bf load_afrag(const bf16* p) {
  v8bf a0 = *reinterpret_cast<const v8bf*>(p);
  v8bf a1 = *reinterpret_cast<const v8bf*>(p + 16);
  v16bf r;
#pragma unroll
  for (int i = 0; i < 8; ++i) { r[i] = a0[i]; r[i + 8] = a1[i]; }
  return r;
}

__global__ __launch_bounds__(256) void wmma_gemm_bt(const bf16* __restrict__ A,
                                                    const bf16* __restrict__ Bt,
                                                    float* __restrict__ C,
                                                    const float* __restrict__ bias,
                                                    int Mgroups, int Ngroups,
                                                    int ldk, int kChunks,
                                                    int ldc, int ncolLimit) {
  int wave = threadIdx.x >> 5;
  int lane = threadIdx.x & 31;
  int grp = blockIdx.x * 8 + wave;
  if (grp >= Mgroups * Ngroups) return;          // uniform per wave: EXEC stays all-1
  int gm = grp / Ngroups;
  int gn = grp - gm * Ngroups;
  int half = lane >> 4;
  int l16 = lane & 15;

  const bf16* arow0 = A + (size_t)(gm * 32 + l16) * ldk + half * 8;
  const bf16* arow1 = arow0 + (size_t)16 * ldk;
  const bf16* br0 = Bt + (size_t)(gn * 64 + l16) * ldk + half * 8;
  const bf16* br1 = br0 + (size_t)16 * ldk;
  const bf16* br2 = br0 + (size_t)32 * ldk;
  const bf16* br3 = br0 + (size_t)48 * ldk;

  v8f acc[2][4];
#pragma unroll
  for (int mi = 0; mi < 2; ++mi)
#pragma unroll
    for (int nj = 0; nj < 4; ++nj)
      acc[mi][nj] = (v8f){0.f, 0.f, 0.f, 0.f, 0.f, 0.f, 0.f, 0.f};

  for (int kc = 0; kc < kChunks; ++kc) {
    int ko = kc * 32;
    __builtin_prefetch(arow0 + ko + 128, 0, 3);     // global_prefetch_b8 ahead
    __builtin_prefetch(br0 + ko + 128, 0, 3);
    v16bf aF[2];
    aF[0] = load_afrag(arow0 + ko);
    aF[1] = load_afrag(arow1 + ko);
    v16bf bF[4];
    bF[0] = load_afrag(br0 + ko);
    bF[1] = load_afrag(br1 + ko);
    bF[2] = load_afrag(br2 + ko);
    bF[3] = load_afrag(br3 + ko);
#pragma unroll
    for (int mi = 0; mi < 2; ++mi)
#pragma unroll
      for (int nj = 0; nj < 4; ++nj)
        acc[mi][nj] = __builtin_amdgcn_wmma_f32_16x16x32_bf16(
            false, aF[mi], false, bF[nj], (short)0, acc[mi][nj], false, false);
  }

#pragma unroll
  for (int nj = 0; nj < 4; ++nj) {
    int col = gn * 64 + nj * 16 + l16;
    if (col < ncolLimit) {
      float badd = bias ? bias[col] : 0.f;
#pragma unroll
      for (int mi = 0; mi < 2; ++mi) {
        int rbase = gm * 32 + mi * 16 + half * 8;
#pragma unroll
        for (int r = 0; r < 8; ++r)
          C[(size_t)(rbase + r) * ldc + col] = acc[mi][nj][r] + badd;
      }
    }
  }
}

// ---------------------------------------------------------------------------
// Scores + softmax + weighted context. One block per point n.
// scores[h][l] = ctx[n,l,:] . qk[h,n,:]   (bk dropped: softmax-invariant)
// ctxw[h][n][c] = sum_l softmax_l(scores)[h][l] * ctx[n][l][c]
// Per-wave dot products reduced with __shfl_xor (wave32).
// ---------------------------------------------------------------------------
__global__ __launch_bounds__(256) void scores_ctxw_kernel(const bf16* __restrict__ ctx,
                                                          const float* __restrict__ qk,
                                                          bf16* __restrict__ actw) {
  __shared__ float sw[NHEAD * LCTX];
  int tid = threadIdx.x;
  int wave = tid >> 5;
  int lane = tid & 31;
  int n = blockIdx.x;
  const bf16* crow = ctx + (size_t)n * LCTX * KPAD;

  for (int p = wave; p < NHEAD * LCTX; p += 8) {
    int h = p >> 4;
    int l = p & 15;
    const float* qrow = qk + ((size_t)h * NPTS + n) * NPADB;
    const bf16* cr = crow + (size_t)l * KPAD;
    float part = 0.f;
    for (int c = lane; c < CFEAT; c += 32) part += bf2f(cr[c]) * qrow[c];
#pragma unroll
    for (int d = 16; d > 0; d >>= 1) part += __shfl_xor(part, d, 32);
    if (lane == 0) sw[p] = part;
  }
  __syncthreads();

  if (tid < NHEAD) {
    int h = tid;
    float mx = sw[h * 16];
#pragma unroll
    for (int l = 1; l < LCTX; ++l) mx = fmaxf(mx, sw[h * 16 + l]);
    float ssum = 0.f;
    float e[LCTX];
#pragma unroll
    for (int l = 0; l < LCTX; ++l) { e[l] = __expf(sw[h * 16 + l] - mx); ssum += e[l]; }
    float inv = 1.f / ssum;
#pragma unroll
    for (int l = 0; l < LCTX; ++l) sw[h * 16 + l] = e[l] * inv;
  }
  __syncthreads();

  for (int c = tid; c < KPAD; c += 256) {
#pragma unroll
    for (int h = 0; h < NHEAD; ++h) {
      float acc = 0.f;
#pragma unroll
      for (int l = 0; l < LCTX; ++l) acc += sw[h * 16 + l] * bf2f(crow[(size_t)l * KPAD + c]);
      actw[((size_t)h * NPTS + n) * KPAD + c] = f2bf(acc);
    }
  }
}

// ---------------------------------------------------------------------------
// Host launcher
// ---------------------------------------------------------------------------
extern "C" void kernel_launch(void* const* d_in, const int* in_sizes, int n_in,
                              void* d_out, int out_size, void* d_ws, size_t ws_size,
                              hipStream_t stream) {
  (void)in_sizes; (void)n_in; (void)out_size; (void)ws_size;
  const float* imgf = (const float*)d_in[0];
  // d_in[1] depths, d_in[4] pointclouds: unused by reference forward
  const float* c2w  = (const float*)d_in[2];
  const float* intr = (const float*)d_in[3];
  const float* qxy  = (const float*)d_in[5];
  const float* qxz  = (const float*)d_in[6];
  const float* qyz  = (const float*)d_in[7];
  const float* Wk   = (const float*)d_in[8];
  const float* Wv   = (const float*)d_in[10];
  const float* bv   = (const float*)d_in[11];
  const float* lng  = (const float*)d_in[12];
  const float* lnb  = (const float*)d_in[13];
  float* out = (float*)d_out;

  char* base = (char*)d_ws;
  size_t off = 0;
  auto take = [&](size_t bytes) -> void* {
    void* p = base + off;
    off = (off + bytes + 255) & ~(size_t)255;
    return p;
  };
  bf16*  wkb   = (bf16*)take((size_t)NHEAD * NPADB * KPAD * 2);
  bf16*  wvb   = (bf16*)take((size_t)NHEAD * NPADB * KPAD * 2);
  float* imgfT = (float*)take((size_t)NVIEW * 4096 * CFEAT * 4);
  bf16*  ctx   = (bf16*)take((size_t)NPTS * LCTX * KPAD * 2);
  bf16*  Aq    = (bf16*)take((size_t)NHEAD * NPTS * KPAD * 2);
  float* qk    = (float*)take((size_t)NHEAD * NPTS * NPADB * 4);
  bf16*  actw  = (bf16*)take((size_t)NHEAD * NPTS * KPAD * 2);

  const int prepBlocks = (NHEAD * NPADB * KPAD + 255) / 256;
  prep_wk_kernel<<<prepBlocks, 256, 0, stream>>>(Wk, wkb);
  prep_wv_kernel<<<prepBlocks, 256, 0, stream>>>(Wv, wvb);
  {
    dim3 tg(4096 / 32, (CFEAT + 31) / 32, NVIEW);   // 128 x 25 x 4
    transpose_imgf_kernel<<<tg, 256, 0, stream>>>(imgf, imgfT);
  }

  const int gemmBlocks = (MGROUPS * NGROUPS + 7) / 8;  // 208

  for (int plane = 0; plane < 3; ++plane) {
    sample_ctx_kernel<<<NPTS * LCTX, 256, 0, stream>>>(imgfT, c2w, intr, ctx, plane);
    query_ln_kernel<<<NPTS, 256, 0, stream>>>(qxy, qxz, qyz, lng, lnb, Aq, plane);

    for (int h = 0; h < NHEAD; ++h)
      wmma_gemm_bt<<<gemmBlocks, 256, 0, stream>>>(
          Aq + (size_t)h * NPTS * KPAD, wkb + (size_t)h * NPADB * KPAD,
          qk + (size_t)h * NPTS * NPADB, nullptr,
          MGROUPS, NGROUPS, KPAD, KCHUNKS, NPADB, NPADB);

    scores_ctxw_kernel<<<NPTS, 256, 0, stream>>>(ctx, qk, actw);

    for (int h = 0; h < NHEAD; ++h)
      wmma_gemm_bt<<<gemmBlocks, 256, 0, stream>>>(
          actw + (size_t)h * NPTS * KPAD, wvb + (size_t)h * NPADB * KPAD,
          out + (size_t)plane * NPTS * QDIM + (size_t)h * CFEAT,
          bv + h * CFEAT,
          MGROUPS, NGROUPS, KPAD, KCHUNKS, QDIM, CFEAT);
  }
}
